// Encoder_31421980738322
// MI455X (gfx1250) — matrix operand order repaired and checked by reading, not compiled, not run on hardware
//
#include <hip/hip_runtime.h>

// MI455X / gfx1250, wave32. Matrix math on v_wmma_f32_16x16x32_bf16;
// attention Q/K tile staging via global_load_async_to_lds_b128 (ASYNCcnt).

typedef __attribute__((ext_vector_type(16))) __bf16 v16bf;
typedef __attribute__((ext_vector_type(4)))  __bf16 bf16x4;
typedef __attribute__((ext_vector_type(8)))  float  v8f;

#define D_MODEL 1024
#define SEQ     1024
#define NBH     128
#define DHEAD   64
#define NEGBIG  -1000000000.0f

// ---------------------------------------------------------------------------
// Async global->LDS copy helpers (gfx1250, tracked by ASYNCcnt)
// ---------------------------------------------------------------------------
__device__ __forceinline__ void async_copy_b128(const void* lds_dst,
                                                unsigned long long gbase,
                                                unsigned byte_off) {
  // Low 32 bits of a flat shared-memory pointer are the LDS byte address
  // (aperture lives in addr[63:32], ISA 10.2).
  const unsigned lds_addr = (unsigned)(uintptr_t)lds_dst;
  asm volatile("global_load_async_to_lds_b128 %0, %1, %2"
               :: "v"(lds_addr), "v"(byte_off), "s"(gbase)
               : "memory");
}

__device__ __forceinline__ void wait_asynccnt0() {
#if defined(__has_builtin) && __has_builtin(__builtin_amdgcn_s_wait_asynccnt)
  __builtin_amdgcn_s_wait_asynccnt(0);
#else
  asm volatile("s_wait_asynccnt 0" ::: "memory");
#endif
}

// ---------------------------------------------------------------------------
// WMMA fragment helpers (ISA 7.12.2 layouts, wave32)
// ---------------------------------------------------------------------------
__device__ __forceinline__ v16bf frag_a(const __bf16* __restrict__ s, int row,
                                        int ld, int kbase, int lane) {
  const int r  = row + (lane & 15);
  const int kb = kbase + ((lane >> 4) << 3);
  const __bf16* p = s + r * ld + kb;
  v16bf v;
#pragma unroll
  for (int i = 0; i < 8; ++i) v[i] = p[i];
#pragma unroll
  for (int i = 0; i < 8; ++i) v[8 + i] = p[16 + i];
  return v;
}

__device__ __forceinline__ v16bf frag_b(const __bf16* __restrict__ sT, int col,
                                        int ld, int kbase, int lane) {
  const int c  = col + (lane & 15);
  const int kb = kbase + ((lane >> 4) << 3);
  const __bf16* p = sT + c * ld + kb;
  v16bf v;
#pragma unroll
  for (int i = 0; i < 8; ++i) v[i] = p[i];
#pragma unroll
  for (int i = 0; i < 8; ++i) v[8 + i] = p[16 + i];
  return v;
}

__device__ __forceinline__ v8f wmma_bf16(v16bf a, v16bf b, v8f c) {
  return __builtin_amdgcn_wmma_f32_16x16x32_bf16(false, a, false, b,
                                                 (short)0, c, false, false);
}

// ---------------------------------------------------------------------------
// Kernel 1: C_bf16[8192,1024] = x_f32 @ W_f32 + bias
// WG = 256 threads = 8 waves, tile 128x128, BK=32 (one WMMA K-step).
// Staging uses float4 (b128) global loads; A-side LDS stores are b64.
// ---------------------------------------------------------------------------
__global__ __launch_bounds__(256, 2)
void qkv_gemm_kernel(const float* __restrict__ A, const float* __restrict__ W,
                     const float* __restrict__ bias, __bf16* __restrict__ C) {
  constexpr int BK = 32, LDA = BK + 8;
  __shared__ __bf16 As[128][LDA];   // [m][k]
  __shared__ __bf16 Ws[128][LDA];   // W tile transposed: [n][k]

  const int tid  = threadIdx.x;
  const int lane = tid & 31;
  const int wave = tid >> 5;
  const int wRow = (wave & 3) * 32;
  const int wCol = (wave >> 2) * 64;
  const int bm   = blockIdx.x * 128;
  const int bn   = blockIdx.y * 128;

  // staging coordinates (vectorized by 4 floats)
  const int ar = tid >> 3, ac = (tid & 7) * 4;    // A tile: 32x(8x4)
  const int wr = tid >> 5, wc = (tid & 31) * 4;   // W tile: 8x(32x4)
  const float* aBase = A + (size_t)(bm + ar) * D_MODEL + ac;
  const float* wBase = W + (size_t)wr * D_MODEL + bn + wc;

  v8f acc[2][4];
#pragma unroll
  for (int mt = 0; mt < 2; ++mt)
#pragma unroll
    for (int nt = 0; nt < 4; ++nt)
#pragma unroll
      for (int r = 0; r < 8; ++r) acc[mt][nt][r] = 0.0f;

  for (int kk = 0; kk < D_MODEL; kk += BK) {
#pragma unroll
    for (int j = 0; j < 4; ++j) {
      const float4 va = *(const float4*)(aBase + (size_t)(32 * j) * D_MODEL + kk);
      bf16x4 pa;
      pa[0] = (__bf16)va.x; pa[1] = (__bf16)va.y;
      pa[2] = (__bf16)va.z; pa[3] = (__bf16)va.w;
      *(bf16x4*)&As[ar + 32 * j][ac] = pa;        // ds_store_b64
    }
#pragma unroll
    for (int j = 0; j < 4; ++j) {
      const float4 vw = *(const float4*)(wBase + (size_t)(kk + 8 * j) * D_MODEL);
      const int kr = wr + 8 * j;
      Ws[wc + 0][kr] = (__bf16)vw.x;
      Ws[wc + 1][kr] = (__bf16)vw.y;
      Ws[wc + 2][kr] = (__bf16)vw.z;
      Ws[wc + 3][kr] = (__bf16)vw.w;
    }
    if (kk + BK < D_MODEL) {  // global_prefetch_b8 for next tiles
      __builtin_prefetch(aBase + kk + BK, 0, 1);
      __builtin_prefetch(wBase + (size_t)(kk + BK) * D_MODEL, 0, 1);
    }
    __syncthreads();

    v16bf af[2], bfr[4];
#pragma unroll
    for (int mt = 0; mt < 2; ++mt)
      af[mt] = frag_a(&As[0][0], wRow + mt * 16, LDA, 0, lane);
#pragma unroll
    for (int nt = 0; nt < 4; ++nt)
      bfr[nt] = frag_b(&Ws[0][0], wCol + nt * 16, LDA, 0, lane);
#pragma unroll
    for (int mt = 0; mt < 2; ++mt)
#pragma unroll
      for (int nt = 0; nt < 4; ++nt)
        acc[mt][nt] = wmma_bf16(af[mt], bfr[nt], acc[mt][nt]);
    __syncthreads();
  }

  const int rh = (lane >> 4) << 3;
  const int ci = lane & 15;
#pragma unroll
  for (int mt = 0; mt < 2; ++mt)
#pragma unroll
    for (int nt = 0; nt < 4; ++nt) {
      const int gc = bn + wCol + nt * 16 + ci;
      const float bv = bias[gc];
#pragma unroll
      for (int r = 0; r < 8; ++r) {
        const int gr = bm + wRow + mt * 16 + rh + r;
        C[(size_t)gr * D_MODEL + gc] = (__bf16)(acc[mt][nt][r] + bv);
      }
    }
}

// ---------------------------------------------------------------------------
// Kernel 2: flash attention over the reshaped [128, 1024, 64] view.
// grid = (8 q-blocks, 128 batches); WG = 256 = 8 waves; wave owns 16 q rows.
// Q/K tiles staged with global_load_async_to_lds_b128; V staged transposed.
// ---------------------------------------------------------------------------
__global__ __launch_bounds__(256, 2)
void attention_kernel(const __bf16* __restrict__ Q, const __bf16* __restrict__ K,
                      const __bf16* __restrict__ V, const int* __restrict__ ds,
                      float* __restrict__ Octx) {
  __shared__ __bf16 Qs[128][72];      // [q][dh]
  __shared__ __bf16 Ks[128][72];      // [key][dh]
  __shared__ __bf16 Vt[DHEAD][136];   // V transposed [dh][key]
  __shared__ __bf16 Ps[8][16][136];   // per-wave P relayout buffer

  const int tid  = threadIdx.x;
  const int lane = tid & 31;
  const int wave = tid >> 5;
  const int bh   = blockIdx.y;
  const int qb   = blockIdx.x * 128;
  const size_t base = (size_t)bh * SEQ * DHEAD;

  // staging coordinates: each thread moves 4 x 16B (8 bf16) chunks per tile
  const int sr = tid >> 3;            // 0..31
  const int sc = (tid & 7) * 8;       // 0,8,..,56
  const unsigned long long qG = (unsigned long long)(uintptr_t)(Q + base);
  const unsigned long long kG = (unsigned long long)(uintptr_t)(K + base);

  // async-stage the Q block (straight copy, bf16, no conversion)
#pragma unroll
  for (int j = 0; j < 4; ++j) {
    const int r = sr + 32 * j;
    async_copy_b128(&Qs[r][sc], qG, (unsigned)(((qb + r) * DHEAD + sc) * 2));
  }

  const int rh = (lane >> 4) << 3;
  const int ci = lane & 15;

  float mrow[8], lrow[8];
  bool  rowmask[8];
  v8f o[4];
#pragma unroll
  for (int r = 0; r < 8; ++r) {
    mrow[r] = -1e30f;
    lrow[r] = 0.0f;
    const int qrow = qb + wave * 16 + rh + r;
    rowmask[r] = (ds[bh * SEQ + qrow] + 1) == 0;
  }
#pragma unroll
  for (int ot = 0; ot < 4; ++ot)
#pragma unroll
    for (int r = 0; r < 8; ++r) o[ot][r] = 0.0f;

  const float scale = 0.125f;  // DH^-0.5

  for (int kb = 0; kb < SEQ; kb += 128) {
    // K block: async straight copy; V block: vectorized load + LDS transpose
#pragma unroll
    for (int j = 0; j < 4; ++j) {
      const int r = sr + 32 * j;
      async_copy_b128(&Ks[r][sc], kG, (unsigned)(((kb + r) * DHEAD + sc) * 2));
      union { uint4 u; __bf16 h[8]; } un;
      un.u = *(const uint4*)(V + base + (size_t)(kb + r) * DHEAD + sc);
#pragma unroll
      for (int i = 0; i < 8; ++i) Vt[sc + i][r] = un.h[i];
    }
    wait_asynccnt0();
    __syncthreads();

    // S[16 x 128] = Q_wave x K^T  (contract dh=64 in 2 WMMA K-steps)
    v8f s[8];
#pragma unroll
    for (int nt = 0; nt < 8; ++nt)
#pragma unroll
      for (int r = 0; r < 8; ++r) s[nt][r] = 0.0f;
#pragma unroll
    for (int kd = 0; kd < DHEAD; kd += 32) {
      const v16bf aq = frag_a(&Qs[0][0], wave * 16, 72, kd, lane);
#pragma unroll
      for (int nt = 0; nt < 8; ++nt) {
        const v16bf bk = frag_b(&Ks[0][0], nt * 16, 72, kd, lane);
        s[nt] = wmma_bf16(aq, bk, s[nt]);
      }
    }

#pragma unroll
    for (int nt = 0; nt < 8; ++nt)
#pragma unroll
      for (int r = 0; r < 8; ++r)
        s[nt][r] = rowmask[r] ? NEGBIG : s[nt][r] * scale;

    // online softmax (row = VGPR idx; 16 lanes of a half hold 16 columns)
#pragma unroll
    for (int r = 0; r < 8; ++r) {
      float mx = s[0][r];
#pragma unroll
      for (int nt = 1; nt < 8; ++nt) mx = fmaxf(mx, s[nt][r]);
#pragma unroll
      for (int m = 8; m >= 1; m >>= 1) mx = fmaxf(mx, __shfl_xor(mx, m, 32));
      const float mnew = fmaxf(mrow[r], mx);
      const float corr = __expf(mrow[r] - mnew);
      mrow[r] = mnew;
      lrow[r] *= corr;
#pragma unroll
      for (int ot = 0; ot < 4; ++ot) o[ot][r] *= corr;
      float rs = 0.0f;
#pragma unroll
      for (int nt = 0; nt < 8; ++nt) {
        const float p = __expf(s[nt][r] - mnew);
        s[nt][r] = p;
        rs += p;
      }
#pragma unroll
      for (int m = 8; m >= 1; m >>= 1) rs += __shfl_xor(rs, m, 32);
      lrow[r] += rs;
    }

    // relayout P (C-layout) -> LDS -> A-fragments
#pragma unroll
    for (int nt = 0; nt < 8; ++nt)
#pragma unroll
      for (int r = 0; r < 8; ++r)
        Ps[wave][rh + r][nt * 16 + ci] = (__bf16)s[nt][r];
    __syncthreads();

    // O += P x V  (contract 128 keys in 4 WMMA K-steps)
#pragma unroll
    for (int kc = 0; kc < 128; kc += 32) {
      const v16bf ap = frag_a(&Ps[wave][0][0], 0, 136, kc, lane);
#pragma unroll
      for (int ot = 0; ot < 4; ++ot) {
        const v16bf bv = frag_b(&Vt[0][0], ot * 16, 136, kc, lane);
        o[ot] = wmma_bf16(ap, bv, o[ot]);
      }
    }
    __syncthreads();   // before next block overwrites Ks/Vt
  }

  // normalize by running sum, write context f32 (flat == [8192,1024])
#pragma unroll
  for (int ot = 0; ot < 4; ++ot)
#pragma unroll
    for (int r = 0; r < 8; ++r) {
      const int qrow = qb + wave * 16 + rh + r;
      const int dcol = ot * 16 + ci;
      Octx[base + (size_t)qrow * DHEAD + dcol] = o[ot][r] / lrow[r];
    }
}

// ---------------------------------------------------------------------------
// Kernel 3: out = LayerNorm(ctx + x) * gamma + beta, one WG per row
// ---------------------------------------------------------------------------
__global__ __launch_bounds__(256)
void add_layernorm_kernel(const float* __restrict__ ctx, const float* __restrict__ x,
                          const float* __restrict__ gamma, const float* __restrict__ beta,
                          float* __restrict__ out) {
  const int row  = blockIdx.x;
  const int tid  = threadIdx.x;
  const int lane = tid & 31;
  const int wave = tid >> 5;
  const float* c  = ctx + (size_t)row * D_MODEL;
  const float* xr = x   + (size_t)row * D_MODEL;

  float v[4], s = 0.0f, s2 = 0.0f;
#pragma unroll
  for (int j = 0; j < 4; ++j) {
    const int idx = tid + 256 * j;
    const float val = c[idx] + xr[idx];
    v[j] = val;
    s += val;
    s2 += val * val;
  }
#pragma unroll
  for (int m = 16; m >= 1; m >>= 1) {
    s  += __shfl_xor(s, m, 32);
    s2 += __shfl_xor(s2, m, 32);
  }
  __shared__ float red[2][8];
  if (lane == 0) { red[0][wave] = s; red[1][wave] = s2; }
  __syncthreads();
  s = 0.0f; s2 = 0.0f;
#pragma unroll
  for (int w = 0; w < 8; ++w) { s += red[0][w]; s2 += red[1][w]; }

  const float mu  = s * (1.0f / D_MODEL);
  const float var = s2 * (1.0f / D_MODEL) - mu * mu;
  const float inv = rsqrtf(var + 1e-5f);
#pragma unroll
  for (int j = 0; j < 4; ++j) {
    const int idx = tid + 256 * j;
    out[(size_t)row * D_MODEL + idx] = (v[j] - mu) * inv * gamma[idx] + beta[idx];
  }
}

// ---------------------------------------------------------------------------
// Launcher. d_ws layout: Q,K,V bf16 [8192*1024] each (48 MB) + ctx f32 (32 MB).
// ---------------------------------------------------------------------------
extern "C" void kernel_launch(void* const* d_in, const int* in_sizes, int n_in,
                              void* d_out, int out_size, void* d_ws, size_t ws_size,
                              hipStream_t stream) {
  const float* x     = (const float*)d_in[0];
  const int*   ds    = (const int*)d_in[1];
  const float* Wq    = (const float*)d_in[2];
  const float* bq    = (const float*)d_in[3];
  const float* Wk    = (const float*)d_in[4];
  const float* bk    = (const float*)d_in[5];
  const float* Wv    = (const float*)d_in[6];
  const float* bv    = (const float*)d_in[7];
  const float* gamma = (const float*)d_in[8];
  const float* beta  = (const float*)d_in[9];
  float* out = (float*)d_out;

  const size_t n = (size_t)8192 * 1024;
  __bf16* Qb = (__bf16*)d_ws;
  __bf16* Kb = Qb + n;
  __bf16* Vb = Kb + n;
  float*  ctx = (float*)(Vb + n);

  dim3 gGemm(64, 8);
  qkv_gemm_kernel<<<gGemm, 256, 0, stream>>>(x, Wq, bq, Qb);
  qkv_gemm_kernel<<<gGemm, 256, 0, stream>>>(x, Wk, bk, Kb);
  qkv_gemm_kernel<<<gGemm, 256, 0, stream>>>(x, Wv, bv, Vb);

  dim3 gAttn(8, NBH);
  attention_kernel<<<gAttn, 256, 0, stream>>>(Qb, Kb, Vb, ds, ctx);

  add_layernorm_kernel<<<8192, 256, 0, stream>>>(ctx, x, gamma, beta, out);
}
